// MultiHeadAttention_68719477245
// MI455X (gfx1250) — matrix operand order repaired and checked by reading, not compiled, not run on hardware
//
#include <hip/hip_runtime.h>
#include <hip/hip_bf16.h>
#include <stdint.h>

// ---------------------------------------------------------------------------
// MHA for MI455X (gfx1250): bf16 WMMA everywhere, fp32 accumulate.
//   H=8, D_IN=E=512, KD=VD=64, B=16, N=1024
// Software-pipelined K-loops so WMMAs overlap the next chunk's b128 loads.
// ---------------------------------------------------------------------------

typedef __bf16 bf16;
typedef __attribute__((ext_vector_type(16))) __bf16 v16bf;
typedef __attribute__((ext_vector_type(8)))  __bf16 v8bf;
typedef __attribute__((ext_vector_type(8)))  float  v8f;

#define HH   8
#define DIN  512
#define EE   512
#define KDIM 64
#define BB   16
#define NN   1024
#define BN   (BB * NN)          // 16384 rows for the big GEMMs

// ---------------- WMMA helpers (layouts per CDNA5 ISA 7.12.2) --------------

static __device__ __forceinline__ v8f wmma_bf16(v16bf a, v16bf b, v8f c) {
  // v_wmma_f32_16x16x32_bf16 : D = A(16x32) * B(32x16) + C(16x16 f32)
  return __builtin_amdgcn_wmma_f32_16x16x32_bf16(
      /*neg_a=*/false, a, /*neg_b=*/false, b,
      /*c_mod=*/(short)0, c, /*reuse_a=*/false, /*reuse_b=*/false);
}

static __device__ __forceinline__ v16bf make_v16(v8bf lo, v8bf hi) {
  v16bf a;
#pragma unroll
  for (int i = 0; i < 8; ++i) { a[i] = lo[i]; a[8 + i] = hi[i]; }
  return a;
}

// A fragment 16x32 (16-bit). row_base = &A[tile_row + (lane&15)][0] (row-major,
// reduction dim contiguous). Lanes<16 hold K = kc+0..7 / kc+16..23,
// lanes>=16 hold K = kc+8..15 / kc+24..31  -> two 16-byte loads.
static __device__ __forceinline__ v16bf load_a_frag(const bf16* row_base,
                                                    int kc, int lane) {
  const bf16* p = row_base + kc + ((lane >> 4) << 3);
  v8bf lo = *(const v8bf*)p;
  v8bf hi = *(const v8bf*)(p + 16);
  return make_v16(lo, hi);
}

// B fragment 32x16 (16-bit). col_base = &Bt[col][0] where Bt is stored with the
// reduction dim contiguous per output column. Lanes<16 hold K = kc+0..7 / kc+16..23,
// lanes>=16 hold K = kc+8..15 / kc+24..31 for column (lane&15) -> two 16-byte loads.
static __device__ __forceinline__ v16bf load_b_frag(const bf16* col_base,
                                                    int kc, int lane) {
  const bf16* p = col_base + kc + ((lane >> 4) << 3);
  v8bf lo = *(const v8bf*)p;
  v8bf hi = *(const v8bf*)(p + 16);
  return make_v16(lo, hi);
}

// ---------------- Stage 0: casts / transposes ------------------------------

__global__ void cvt_f32_bf16(const float* __restrict__ in,
                             bf16* __restrict__ out, int n) {
  int i = blockIdx.x * blockDim.x + threadIdx.x;
  if (i < n) out[i] = (bf16)in[i];
}

// W:(H, DIN, 64) fp32  ->  Wt[(h*64+d)][DIN] bf16
__global__ void cvt_w_qkv(const float* __restrict__ W, bf16* __restrict__ Wt) {
  int i = blockIdx.x * blockDim.x + threadIdx.x;   // over H*DIN*64
  int kd  = i & 63;
  int din = (i >> 6) & (DIN - 1);
  int h   = i >> 15;
  Wt[((h << 6) + kd) * DIN + din] = (bf16)W[i];
}

// W_out:(H, 64, EE) fp32 -> Wt[e][(h*64+v)] bf16
__global__ void cvt_w_out(const float* __restrict__ W, bf16* __restrict__ Wt) {
  int i = blockIdx.x * blockDim.x + threadIdx.x;   // over H*64*EE
  int e  = i & (EE - 1);
  int hv = i >> 9;                                 // 0..511
  Wt[e * (HH * KDIM) + hv] = (bf16)W[i];
}

// ---------------- Stage 1: Q/K/V projection GEMMs --------------------------
// C(16384 x 512) = qbf(16384 x 512) * Wt^T ; wave computes a 16x64 tile.
// Double-buffered K-loop: next chunk's fragments load while current WMMAs run.
// mode 0/1: write Out[h][b][n][kd] (row-major kd)   (Q and K)
// mode 2  : write Out[h][b][vd][n] (transposed)     (V, for P*V B-fragments)
__global__ void proj_gemm(const bf16* __restrict__ A, const bf16* __restrict__ Wt,
                          bf16* __restrict__ Out, int mode) {
  int lane = threadIdx.x & 31;
  int wid  = blockIdx.x * (blockDim.x >> 5) + (threadIdx.x >> 5);
  int r0   = (wid >> 3) << 4;     // 1024 row tiles
  int c0   = (wid & 7)  << 6;     // 8 groups of 64 cols
  int half = lane >> 4;
  int lr   = lane & 15;

  const bf16* arow = A + (r0 + lr) * DIN;
  const bf16* bcol[4];
#pragma unroll
  for (int j = 0; j < 4; ++j) bcol[j] = Wt + (c0 + j * 16 + lr) * DIN;

  v8f acc[4] = {};
  v16bf a_cur = load_a_frag(arow, 0, lane);
  v16bf b_cur[4];
#pragma unroll
  for (int j = 0; j < 4; ++j) b_cur[j] = load_b_frag(bcol[j], 0, lane);

  for (int kc = 0; kc < DIN - 32; kc += 32) {
    // prefetch next chunk into registers (overlaps the 4 WMMAs below)
    v16bf a_nxt = load_a_frag(arow, kc + 32, lane);
    v16bf b_nxt[4];
#pragma unroll
    for (int j = 0; j < 4; ++j) b_nxt[j] = load_b_frag(bcol[j], kc + 32, lane);
#pragma unroll
    for (int j = 0; j < 4; ++j) acc[j] = wmma_bf16(a_cur, b_cur[j], acc[j]);
    a_cur = a_nxt;
#pragma unroll
    for (int j = 0; j < 4; ++j) b_cur[j] = b_nxt[j];
  }
#pragma unroll
  for (int j = 0; j < 4; ++j) acc[j] = wmma_bf16(a_cur, b_cur[j], acc[j]);

#pragma unroll
  for (int j = 0; j < 4; ++j) {
    int hk = c0 + j * 16 + lr;
    int h  = hk >> 6, kd = hk & 63;
#pragma unroll
    for (int r = 0; r < 8; ++r) {
      int bn = r0 + r + 8 * half;
      int b_ = bn >> 10, n = bn & (NN - 1);
      bf16 v = (bf16)acc[j][r];
      if (mode == 2)
        Out[((h * BB + b_) * KDIM + kd) * NN + n] = v;       // Vt[h][b][vd][n]
      else
        Out[((h * BB + b_) * NN + n) * KDIM + kd] = v;       // [h][b][n][kd]
    }
  }
}

// ---------------- Stage 2: fused flash attention ---------------------------
// One wave per (h, b, 16-row query tile).  Online softmax, O accumulated fp32.
// K/V fragments for a 32-key step are issued up-front; V load latency hides
// behind the softmax VALU/trans stretch. Next step prefetched to L2.
__global__ void attn_kernel(const bf16* __restrict__ Qb, const bf16* __restrict__ Kb,
                            const bf16* __restrict__ Vt, const int* __restrict__ mask,
                            bf16* __restrict__ Hcat) {
  __shared__ bf16 lds[4][16 * 32];
  int lane = threadIdx.x & 31;
  int warp = threadIdx.x >> 5;
  int wid  = blockIdx.x * (blockDim.x >> 5) + warp;
  int h    = wid / (BB * 64);
  int rem  = wid % (BB * 64);
  int b    = rem >> 6;
  int n0   = (rem & 63) << 4;
  int half = lane >> 4;
  int lr   = lane & 15;

  const bf16* Qhb = Qb + (h * BB + b) * NN * KDIM;
  const bf16* Khb = Kb + (h * BB + b) * NN * KDIM;
  const bf16* Vhb = Vt + (h * BB + b) * KDIM * NN;
  const int*  Mrow = mask + (b * NN + n0) * NN;
  bf16* P = &lds[warp][0];

  const bf16* vcol[4];
#pragma unroll
  for (int j = 0; j < 4; ++j) vcol[j] = Vhb + (j * 16 + lr) * NN;

  // Q fragments for the whole tile (kd = 64 -> 2 chunks), kept in registers.
  v16bf aq0 = load_a_frag(Qhb + (n0 + lr) * KDIM, 0,  lane);
  v16bf aq1 = load_a_frag(Qhb + (n0 + lr) * KDIM, 32, lane);

  float mrun[8], lrun[8];
  v8f O[4] = {};
#pragma unroll
  for (int r = 0; r < 8; ++r) { mrun[r] = -3.0e38f; lrun[r] = 0.0f; }

  for (int g0 = 0; g0 < NN; g0 += 32) {
    // ---- issue all K and V fragment loads for this 32-key step ----
    const bf16* kcol0 = Khb + (g0 + lr) * KDIM;
    const bf16* kcol1 = Khb + (g0 + 16 + lr) * KDIM;
    v16bf bk[4], bv[4];
    bk[0] = load_b_frag(kcol0, 0,  lane);
    bk[1] = load_b_frag(kcol0, 32, lane);
    bk[2] = load_b_frag(kcol1, 0,  lane);
    bk[3] = load_b_frag(kcol1, 32, lane);
#pragma unroll
    for (int j = 0; j < 4; ++j) bv[j] = load_b_frag(vcol[j], g0, lane);
    if (g0 + 32 < NN) {       // L2 prefetch of next step (speculative-safe)
      __builtin_prefetch(Khb + (g0 + 32 + lr) * KDIM, 0, 1);
      __builtin_prefetch(Khb + (g0 + 48 + lr) * KDIM, 0, 1);
      __builtin_prefetch(vcol[0] + g0 + 32, 0, 1);
      __builtin_prefetch(vcol[2] + g0 + 32, 0, 1);
    }
    // ---- scores S = (1/8) Q K^T for 32 keys (two 16x16 C tiles) ----
    v8f s[2];
#pragma unroll
    for (int t = 0; t < 2; ++t) {
      v8f c = {};
      c = wmma_bf16(aq0, bk[2 * t + 0], c);
      c = wmma_bf16(aq1, bk[2 * t + 1], c);
      s[t] = c;
    }
    // ---- scale + mask ----
    float sv[2][8];
#pragma unroll
    for (int t = 0; t < 2; ++t) {
      int g = g0 + t * 16 + lr;
#pragma unroll
      for (int r = 0; r < 8; ++r) {
        int row = r + 8 * half;
        int mm  = Mrow[row * NN + g];
        sv[t][r] = mm ? -1.0e30f : 0.125f * s[t][r];
      }
    }
    // ---- online softmax: row max / exp / row sum (butterflies in half-wave) ----
    float mnew[8], alpha[8];
#pragma unroll
    for (int r = 0; r < 8; ++r) {
      float v = fmaxf(sv[0][r], sv[1][r]);
      v = fmaxf(v, __shfl_xor(v, 1, 32));
      v = fmaxf(v, __shfl_xor(v, 2, 32));
      v = fmaxf(v, __shfl_xor(v, 4, 32));
      v = fmaxf(v, __shfl_xor(v, 8, 32));
      mnew[r]  = fmaxf(mrun[r], v);
      alpha[r] = __expf(mrun[r] - mnew[r]);
      mrun[r]  = mnew[r];
    }
#pragma unroll
    for (int r = 0; r < 8; ++r) {
      float p0 = __expf(sv[0][r] - mnew[r]);
      float p1 = __expf(sv[1][r] - mnew[r]);
      float rs = p0 + p1;
      rs += __shfl_xor(rs, 1, 32);
      rs += __shfl_xor(rs, 2, 32);
      rs += __shfl_xor(rs, 4, 32);
      rs += __shfl_xor(rs, 8, 32);
      lrun[r] = lrun[r] * alpha[r] + rs;
      int row = r + 8 * half;
      P[row * 32 + lr]      = (bf16)p0;   // C-layout -> row-major P in LDS
      P[row * 32 + 16 + lr] = (bf16)p1;
#pragma unroll
      for (int j = 0; j < 4; ++j) O[j][r] *= alpha[r];
    }
    asm volatile("s_wait_dscnt 0x0" ::: "memory");   // P stores visible in-wave
    // ---- O += P * V  (P as A-fragment from LDS, V fragments already here) ----
    v16bf ap;
    {
      const bf16* pp = P + lr * 32 + ((lane >> 4) << 3);
      v8bf lo = *(const v8bf*)pp;
      v8bf hi = *(const v8bf*)(pp + 16);
      ap = make_v16(lo, hi);
    }
#pragma unroll
    for (int j = 0; j < 4; ++j) O[j] = wmma_bf16(ap, bv[j], O[j]);
  }
  // ---- normalize and emit heads concatenated: Hcat[bn][h*64+v] (bf16) ----
#pragma unroll
  for (int j = 0; j < 4; ++j) {
    int hv = (h << 6) + (j << 4) + lr;
#pragma unroll
    for (int r = 0; r < 8; ++r) {
      int bn = b * NN + n0 + r + 8 * half;
      Hcat[bn * (HH * KDIM) + hv] = (bf16)(O[j][r] / lrun[r]);
    }
  }
}

// ---------------- Stage 3: output projection -------------------------------
// out(16384 x 512) = Hcat(16384 x 512) * W_out ; fp32 result. Same pipelining.
__global__ void out_gemm(const bf16* __restrict__ A, const bf16* __restrict__ Wt,
                         float* __restrict__ Out) {
  int lane = threadIdx.x & 31;
  int wid  = blockIdx.x * (blockDim.x >> 5) + (threadIdx.x >> 5);
  int r0   = (wid >> 3) << 4;
  int c0   = (wid & 7)  << 6;
  int half = lane >> 4;
  int lr   = lane & 15;

  const bf16* arow = A + (r0 + lr) * (HH * KDIM);
  const bf16* bcol[4];
#pragma unroll
  for (int j = 0; j < 4; ++j) bcol[j] = Wt + (c0 + j * 16 + lr) * (HH * KDIM);

  v8f acc[4] = {};
  v16bf a_cur = load_a_frag(arow, 0, lane);
  v16bf b_cur[4];
#pragma unroll
  for (int j = 0; j < 4; ++j) b_cur[j] = load_b_frag(bcol[j], 0, lane);

  for (int kc = 0; kc < HH * KDIM - 32; kc += 32) {
    v16bf a_nxt = load_a_frag(arow, kc + 32, lane);
    v16bf b_nxt[4];
#pragma unroll
    for (int j = 0; j < 4; ++j) b_nxt[j] = load_b_frag(bcol[j], kc + 32, lane);
#pragma unroll
    for (int j = 0; j < 4; ++j) acc[j] = wmma_bf16(a_cur, b_cur[j], acc[j]);
    a_cur = a_nxt;
#pragma unroll
    for (int j = 0; j < 4; ++j) b_cur[j] = b_nxt[j];
  }
#pragma unroll
  for (int j = 0; j < 4; ++j) acc[j] = wmma_bf16(a_cur, b_cur[j], acc[j]);

#pragma unroll
  for (int j = 0; j < 4; ++j) {
    int e = c0 + j * 16 + lr;
#pragma unroll
    for (int r = 0; r < 8; ++r) {
      int bn = r0 + r + 8 * half;
      Out[bn * EE + e] = acc[j][r];
    }
  }
}

// ---------------- host launch ----------------------------------------------

extern "C" void kernel_launch(void* const* d_in, const int* in_sizes, int n_in,
                              void* d_out, int out_size, void* d_ws, size_t ws_size,
                              hipStream_t stream) {
  const float* q    = (const float*)d_in[0];
  const int*   mask = (const int*)d_in[1];      // bool -> int per harness contract
  const float* Wq   = (const float*)d_in[2];
  const float* Wk   = (const float*)d_in[3];
  const float* Wv   = (const float*)d_in[4];
  const float* Wo   = (const float*)d_in[5];
  float* out = (float*)d_out;

  // workspace layout (bf16 buffers, all offsets 256B-aligned)
  char* ws = (char*)d_ws;
  size_t off = 0;
  bf16* qbf  = (bf16*)(ws + off); off += (size_t)BN * DIN * 2;            // 16 MB
  bf16* Wqt  = (bf16*)(ws + off); off += (size_t)512 * 512 * 2;
  bf16* Wkt  = (bf16*)(ws + off); off += (size_t)512 * 512 * 2;
  bf16* Wvt  = (bf16*)(ws + off); off += (size_t)512 * 512 * 2;
  bf16* Wot  = (bf16*)(ws + off); off += (size_t)512 * 512 * 2;
  bf16* Qb   = (bf16*)(ws + off); off += (size_t)HH * BB * NN * KDIM * 2; // 16 MB
  bf16* Kb   = (bf16*)(ws + off); off += (size_t)HH * BB * NN * KDIM * 2; // 16 MB
  bf16* Vt   = (bf16*)(ws + off); off += (size_t)HH * BB * NN * KDIM * 2; // 16 MB
  bf16* Hcat = (bf16*)(ws + off); off += (size_t)BN * HH * KDIM * 2;      // 16 MB

  // Stage 0: casts + weight transposes
  cvt_f32_bf16<<<(BN * DIN + 255) / 256, 256, 0, stream>>>(q, qbf, BN * DIN);
  cvt_w_qkv<<<(HH * DIN * KDIM + 255) / 256, 256, 0, stream>>>(Wq, Wqt);
  cvt_w_qkv<<<(HH * DIN * KDIM + 255) / 256, 256, 0, stream>>>(Wk, Wkt);
  cvt_w_qkv<<<(HH * DIN * KDIM + 255) / 256, 256, 0, stream>>>(Wv, Wvt);
  cvt_w_out<<<(HH * KDIM * EE + 255) / 256, 256, 0, stream>>>(Wo, Wot);

  // Stage 1: projections (8192 waves each = 2048 blocks x 4 waves)
  proj_gemm<<<2048, 128, 0, stream>>>(qbf, Wqt, Qb, 0);
  proj_gemm<<<2048, 128, 0, stream>>>(qbf, Wkt, Kb, 1);
  proj_gemm<<<2048, 128, 0, stream>>>(qbf, Wvt, Vt, 2);

  // Stage 2: fused attention (H*B*64 = 8192 waves)
  attn_kernel<<<2048, 128, 0, stream>>>(Qb, Kb, Vt, mask, Hcat);

  // Stage 3: output projection
  out_gemm<<<2048, 128, 0, stream>>>(Hcat, Wot, out);
}
